// GraphVAEEncoder_26723286515960
// MI455X (gfx1250) — compile-verified, matrix operand order checked by reading, not run on hardware
//
#include <hip/hip_runtime.h>
#include <math.h>

#define NN 100000
#define EE 1600000
#define EA_ (EE + NN)
#define FN 64
#define FE 16
#define HID_ 128
#define HEADS_ 4
#define OUT_ 32
#define LL 3
#define BB 64
#define NEG 0.2f

typedef __attribute__((ext_vector_type(16))) __bf16 v16bf;
typedef __attribute__((ext_vector_type(8)))  float  v8f;
typedef __attribute__((ext_vector_type(4)))  unsigned int v4u32;
typedef __attribute__((ext_vector_type(8)))  int v8i32;
typedef __attribute__((ext_vector_type(4)))  int v4i32;

__device__ __forceinline__ unsigned short f2bf(float f) {
  unsigned int u = __float_as_uint(f);
  unsigned int r = (u + 0x7FFFu + ((u >> 16) & 1u)) >> 16;  // RNE
  return (unsigned short)r;
}

union FragU { v16bf v; unsigned short s[16]; unsigned int d[8]; };

// fp32 source -> bf16 fragment (only tiny VAE-head kernel uses this now).
// CDNA5 ISA 7.12.2 16-bit operand layout: lane L holds row/col (L&15);
// K striping: +8 for lanes 16-31, VGPR pairs cover K {0..15} then {16..31}.
__device__ __forceinline__ v16bf load_frag_row(const float* __restrict__ row, int lane,
                                               int k0, int kmax) {
  FragU u;
#pragma unroll
  for (int e = 0; e < 16; ++e) {
    int vr = e >> 1;
    int k = ((vr < 4) ? 0 : 16) + ((lane >> 4) << 3) + ((vr & 3) << 1) + (e & 1) + k0;
    float f = (k < kmax) ? row[k] : 0.0f;
    u.s[e] = f2bf(f);
  }
  return u.v;
}

// bf16 source -> fragment: K pairs are contiguous, so 8 dword loads, zero VALU convert.
__device__ __forceinline__ v16bf load_frag_bf16(const unsigned short* __restrict__ row,
                                                int lane, int k0) {
  FragU u;
#pragma unroll
  for (int j = 0; j < 8; ++j) {
    int k = ((j < 4) ? 0 : 16) + ((lane >> 4) << 3) + ((j & 3) << 1) + k0;
    u.d[j] = *(const unsigned int*)(const void*)(row + k);
  }
  return u.v;
}

// bf16 source with K bound (zero-fill K >= kmax; used for the K=16 edge operands).
__device__ __forceinline__ v16bf load_frag_bf16k(const unsigned short* __restrict__ row,
                                                 int lane, int kmax) {
  FragU u;
#pragma unroll
  for (int j = 0; j < 8; ++j) {
    int k = ((j < 4) ? 0 : 16) + ((lane >> 4) << 3) + ((j & 3) << 1);
    u.d[j] = (k < kmax) ? *(const unsigned int*)(const void*)(row + k) : 0u;
  }
  return u.v;
}

__device__ __forceinline__ v8f wmma_bf16(v16bf a, v16bf b, v8f c) {
  return __builtin_amdgcn_wmma_f32_16x16x32_bf16(false, a, false, b, (short)0, c,
                                                 false, false);
}

// ---------------------------------------------------------------- fill
__global__ void k_fill(float* __restrict__ p, float v, size_t n) {
  size_t t = (size_t)blockIdx.x * blockDim.x + threadIdx.x;
  if (t < n) p[t] = v;
}

// -------------------------------------------------- fp32 -> bf16 mirror
__global__ void k_w2bf(const float* __restrict__ W, unsigned short* __restrict__ o, int n) {
  int t = blockIdx.x * blockDim.x + threadIdx.x;
  if (t < n) o[t] = f2bf(W[t]);
}

// ------------------------------------------- fold attention vectors into W
// weff[((l*3+which)*4+h)*128+k] = sum_o att[l,h,o] * W[l, h*32+o, k]
__global__ void k_weff(const float* __restrict__ gat_W, const float* __restrict__ gat_We,
                       const float* __restrict__ att_src, const float* __restrict__ att_dst,
                       const float* __restrict__ att_edge, float* __restrict__ weff) {
  int t = blockIdx.x * blockDim.x + threadIdx.x;
  if (t >= LL * 3 * HEADS_ * HID_) return;
  int k = t & 127;
  int r = t >> 7;
  int hh = r & 3; r >>= 2;
  int which = r % 3, l = r / 3;
  const float* W = (which == 2 ? gat_We : gat_W) + (size_t)l * HID_ * HID_;
  const float* att = (which == 0 ? att_src : which == 1 ? att_dst : att_edge)
                     + ((size_t)l * HEADS_ + hh) * OUT_;
  float s = 0.f;
#pragma unroll
  for (int o = 0; o < OUT_; ++o) s += att[o] * W[(size_t)(hh * OUT_ + o) * HID_ + k];
  weff[t] = s;
}

// ------------------------------------------------- node encoder (WMMA bf16)
__global__ __launch_bounds__(256) void k_node_enc(const unsigned short* __restrict__ xbf,
                                                  const unsigned short* __restrict__ wnbf,
                                                  const float* __restrict__ bn,
                                                  float* __restrict__ h,
                                                  unsigned short* __restrict__ hbf) {
  int lane = threadIdx.x & 31, wave = threadIdx.x >> 5;
  int r0 = blockIdx.x * 16, m = lane & 15, c0 = wave * 16;
  v8f acc = {};
#pragma unroll
  for (int t = 0; t < 2; ++t) {
    v16bf a = load_frag_bf16(xbf + (size_t)(r0 + m) * FN, lane, 32 * t);
    v16bf b = load_frag_bf16(wnbf + (size_t)(c0 + m) * FN, lane, 32 * t);
    acc = wmma_bf16(a, b, acc);
  }
  int nrel = lane & 15, mhi = (lane >> 4) * 8;
#pragma unroll
  for (int r = 0; r < 8; ++r) {
    int col = c0 + nrel;
    float v = acc[r] + bn[col];
    v = v > 0.f ? v : 0.f;
    size_t idx = (size_t)(r0 + r + mhi) * HID_ + col;
    h[idx] = v;
    hbf[idx] = f2bf(v);
  }
}

// --- edge encoder fused: eh=relu(ea@WeT+be); a_e[l][e][h]; loop_sum/deg atomics.
__global__ __launch_bounds__(256) void k_edge_enc(const unsigned short* __restrict__ eabf,
                                                  const int* __restrict__ ei,
                                                  const unsigned short* __restrict__ webf,
                                                  const float* __restrict__ be,
                                                  const float* __restrict__ weff,
                                                  float* __restrict__ a_e,
                                                  float* __restrict__ loop_sum,
                                                  float* __restrict__ deg) {
  __shared__ float tile[16][HID_];
  __shared__ int sdst[16];
  int lane = threadIdx.x & 31, wave = threadIdx.x >> 5;
  int e0 = blockIdx.x * 16;
  if (threadIdx.x < 16) sdst[threadIdx.x] = ei[EE + e0 + threadIdx.x];
  int m = lane & 15, c0 = wave * 16;
  v8f acc = {};
  v16bf a = load_frag_bf16k(eabf + (size_t)(e0 + m) * FE, lane, FE);  // K=16, zero-pad
  v16bf b = load_frag_bf16k(webf + (size_t)(c0 + m) * FE, lane, FE);
  acc = wmma_bf16(a, b, acc);
  int nrel = lane & 15, mhi = (lane >> 4) * 8;
#pragma unroll
  for (int r = 0; r < 8; ++r) {
    int col = c0 + nrel;
    float v = acc[r] + be[col];
    tile[r + mhi][col] = v > 0.f ? v : 0.f;
  }
  __syncthreads();
  if (threadIdx.x < 192) {  // 16 edges x (3 layers x 4 heads)
    int mm = threadIdx.x / 12, q = threadIdx.x % 12;
    int l = q >> 2, hh = q & 3;
    const float* w = weff + ((size_t)(l * 3 + 2) * HEADS_ + hh) * HID_;
    float s = 0.f;
#pragma unroll 8
    for (int k = 0; k < HID_; ++k) s += tile[mm][k] * w[k];
    a_e[((size_t)l * EA_ + e0 + mm) * HEADS_ + hh] = s;
  }
  if (threadIdx.x < 16) atomicAdd(&deg[sdst[threadIdx.x]], 1.0f);
#pragma unroll
  for (int i = 0; i < 8; ++i) {
    int idx = threadIdx.x + 256 * i;
    int mm = idx >> 7, c = idx & 127;
    atomicAdd(&loop_sum[(size_t)sdst[mm] * HID_ + c], tile[mm][c]);
  }
}

// ----------- self-loop edge attr: mean of incoming eh, folded into a_e[l][E+n]
__global__ void k_selfloop(const float* __restrict__ loop_sum, const float* __restrict__ deg,
                           const float* __restrict__ weff, float* __restrict__ a_e) {
  int t = blockIdx.x * blockDim.x + threadIdx.x;
  if (t >= NN * 12) return;
  int n = t / 12, q = t % 12, l = q >> 2, hh = q & 3;
  float d = deg[n]; d = d > 1.f ? d : 1.f;
  const float* row = loop_sum + (size_t)n * HID_;
  const float* w = weff + ((size_t)(l * 3 + 2) * HEADS_ + hh) * HID_;
  float s = 0.f;
#pragma unroll 8
  for (int k = 0; k < HID_; ++k) s += row[k] * w[k];
  a_e[((size_t)l * EA_ + EE + n) * HEADS_ + hh] = s / d;
}

// ------- per layer: xh = h @ Wl^T. Weights staged to LDS via Tensor Data Mover,
// A operand streamed from bf16 mirror (no convert VALU). Folded a_src/a_dst dots.
__global__ __launch_bounds__(256) void k_xh_attn(const unsigned short* __restrict__ hbf,
                                                 const unsigned short* __restrict__ Wbf,
                                                 const float* __restrict__ weff, int l,
                                                 float* __restrict__ xh,
                                                 float* __restrict__ a_s,
                                                 float* __restrict__ a_d) {
  __shared__ unsigned short sW[HID_ * HID_];  // 32 KB weight tile
  __shared__ float tile[16][HID_];
  int lane = threadIdx.x & 31, wave = threadIdx.x >> 5;
  int r0 = blockIdx.x * 16, m = lane & 15, c0 = wave * 16;

  if (wave == 0) {
    // Tensor DMA: 2D tensor, 128x128 elements of 2 bytes, tile == tensor.
    unsigned long long ga = (unsigned long long)(uintptr_t)Wbf;
    unsigned int lds = (unsigned int)(uintptr_t)(&sW[0]);  // low 32 bits = LDS offset
    v4u32 g0;
    g0[0] = 1u;                                   // count=1, user descriptor
    g0[1] = lds;                                  // lds_addr
    g0[2] = (unsigned int)(ga & 0xFFFFFFFFu);     // global_addr[31:0]
    g0[3] = (unsigned int)((ga >> 32) & 0x1FFFFFFu) | (2u << 30);  // addr hi | type=2
    v8i32 g1;
    g1[0] = (int)(1u << 16);          // workgroup_mask=0, data_size=1 (2 bytes)
    g1[1] = (int)(128u << 16);        // tensor_dim0 = 128 (low 16 bits)
    g1[2] = (int)(128u << 16);        // tensor_dim0 hi=0 | tensor_dim1 = 128 (low)
    g1[3] = (int)(128u << 16);        // tensor_dim1 hi=0 | tile_dim0 = 128
    g1[4] = (int)128u;                // tile_dim1 = 128, tile_dim2 = 0
    g1[5] = (int)128u;                // tensor_dim0_stride[31:0] = 128
    g1[6] = 0;                        // stride0 hi | stride1 lo
    g1[7] = 0;                        // stride1 hi
    v4i32 g2 = {0, 0, 0, 0};
    v4i32 g3 = {0, 0, 0, 0};
    v8i32 g4 = {0, 0, 0, 0, 0, 0, 0, 0};
    __builtin_amdgcn_tensor_load_to_lds(g0, g1, g2, g3, g4, 0);
    __builtin_amdgcn_s_wait_tensorcnt(0);
  }
  __syncthreads();

  v8f acc = {};
#pragma unroll
  for (int t = 0; t < 4; ++t) {
    v16bf a = load_frag_bf16(hbf + (size_t)(r0 + m) * HID_, lane, 32 * t);
    v16bf b = load_frag_bf16(&sW[(c0 + m) * HID_], lane, 32 * t);
    acc = wmma_bf16(a, b, acc);
  }
  int nrel = lane & 15, mhi = (lane >> 4) * 8;
#pragma unroll
  for (int r = 0; r < 8; ++r) {
    int col = c0 + nrel;
    float v = acc[r];
    tile[r + mhi][col] = v;
    xh[(size_t)(r0 + r + mhi) * HID_ + col] = v;
  }
  __syncthreads();
  if (threadIdx.x < 128) {
    int mm = threadIdx.x >> 3, q = threadIdx.x & 7;
    int which = q >> 2, hh = q & 3;
    const float* w = weff + ((size_t)(l * 3 + which) * HEADS_ + hh) * HID_;
    float s = 0.f;
#pragma unroll 8
    for (int k = 0; k < HID_; ++k) s += tile[mm][k] * w[k];
    (which ? a_d : a_s)[(size_t)(blockIdx.x * 16 + mm) * HEADS_ + hh] = s;
  }
}

// ------------------------- segment softmax pass 1: logits + atomic segment max
__global__ void k_logit_max(const int* __restrict__ ei, const float* __restrict__ a_s,
                            const float* __restrict__ a_d, const float* __restrict__ a_e,
                            int l, float* __restrict__ logit, float* __restrict__ segmax) {
  size_t t = (size_t)blockIdx.x * blockDim.x + threadIdx.x;
  if (t >= (size_t)EA_ * HEADS_) return;
  int a = (int)(t >> 2), hh = (int)(t & 3);
  int srcn, dstn;
  if (a < EE) { srcn = ei[a]; dstn = ei[EE + a]; } else { srcn = dstn = a - EE; }
  float lg = a_s[(size_t)srcn * HEADS_ + hh] + a_d[(size_t)dstn * HEADS_ + hh]
           + a_e[((size_t)l * EA_ + a) * HEADS_ + hh];
  lg = lg > 0.f ? lg : NEG * lg;  // leaky_relu
  logit[t] = lg;
  atomicMax(&segmax[(size_t)dstn * HEADS_ + hh], lg);
}

// ------------------------- segment softmax pass 2: exp + atomic segment sum
__global__ void k_exp_sum(const int* __restrict__ ei, float* __restrict__ logit,
                          const float* __restrict__ segmax, float* __restrict__ segsum,
                          const float* __restrict__ xh) {
  size_t t = (size_t)blockIdx.x * blockDim.x + threadIdx.x;
  if (t >= (size_t)EA_ * HEADS_) return;
  int a = (int)(t >> 2), hh = (int)(t & 3);
  int dstn = (a < EE) ? ei[EE + a] : a - EE;
  float e = __expf(logit[t] - segmax[(size_t)dstn * HEADS_ + hh]);
  logit[t] = e;  // buffer now holds un-normalized alpha
  atomicAdd(&segsum[(size_t)dstn * HEADS_ + hh], e);
  if (hh == 0) {
    int srcn = (a < EE) ? ei[a] : a - EE;
    __builtin_prefetch(xh + (size_t)srcn * HID_, 0, 3);  // global_prefetch_b8
  }
}

// ------------------------- message scatter: hacc[dst] += xh[src] * alpha_norm
__global__ __launch_bounds__(128) void k_scatter(const int* __restrict__ ei,
                                                 const float* __restrict__ alpha,
                                                 const float* __restrict__ segsum,
                                                 const float* __restrict__ xh,
                                                 float* __restrict__ hacc) {
  __shared__ float w[HEADS_];
  int a = blockIdx.x;
  int srcn, dstn;
  if (a < EE) { srcn = ei[a]; dstn = ei[EE + a]; } else { srcn = dstn = a - EE; }
  if (threadIdx.x < HEADS_)
    w[threadIdx.x] = alpha[(size_t)a * HEADS_ + threadIdx.x]
                   / (segsum[(size_t)dstn * HEADS_ + threadIdx.x] + 1e-16f);
  __syncthreads();
  int c = threadIdx.x;
  float v = xh[(size_t)srcn * HID_ + c] * w[c >> 5];
  atomicAdd(&hacc[(size_t)dstn * HID_ + c], v);
}

__global__ void k_finish(const float* __restrict__ hacc, const float* __restrict__ gat_b,
                         int l, float* __restrict__ h, unsigned short* __restrict__ hbf) {
  size_t t = (size_t)blockIdx.x * blockDim.x + threadIdx.x;
  if (t >= (size_t)NN * HID_) return;
  float v = hacc[t] + gat_b[l * HID_ + (t & 127)];
  v = v > 0.f ? v : 0.f;
  h[t] = v;
  hbf[t] = f2bf(v);
}

__global__ __launch_bounds__(128) void k_pool(const float* __restrict__ h,
                                              const int* __restrict__ batch,
                                              float* __restrict__ pooled,
                                              float* __restrict__ cnt) {
  int n = blockIdx.x, c = threadIdx.x;
  int b = batch[n];
  atomicAdd(&pooled[(size_t)b * HID_ + c], h[(size_t)n * HID_ + c]);
  if (c == 0) atomicAdd(&cnt[b], 1.0f);
}

__global__ void k_pnorm(const float* __restrict__ pooled, const float* __restrict__ cnt,
                        float* __restrict__ pnorm) {
  int t = blockIdx.x * blockDim.x + threadIdx.x;
  if (t >= BB * HID_) return;
  float c = cnt[t >> 7]; c = c > 1.f ? c : 1.f;
  pnorm[t] = pooled[t] / c;
}

// ----------------------------------- VAE heads: mu / logvar (WMMA, K=128)
__global__ __launch_bounds__(256) void k_head(const float* __restrict__ pnorm,
                                              const float* __restrict__ Wmu,
                                              const float* __restrict__ bmu,
                                              const float* __restrict__ Wlv,
                                              const float* __restrict__ blv,
                                              float* __restrict__ out) {
  int lane = threadIdx.x & 31, wave = threadIdx.x >> 5;
  int mat = blockIdx.x >> 2, rt = blockIdx.x & 3;
  const float* W = mat ? Wlv : Wmu;
  const float* bias = mat ? blv : bmu;
  int r0 = rt * 16, c0 = wave * 16, m = lane & 15;
  v8f acc = {};
#pragma unroll
  for (int t = 0; t < 4; ++t) {
    v16bf a = load_frag_row(pnorm + (size_t)(r0 + m) * HID_, lane, 32 * t, HID_);
    v16bf b = load_frag_row(W + (size_t)(c0 + m) * HID_, lane, 32 * t, HID_);
    acc = wmma_bf16(a, b, acc);
  }
  int nrel = lane & 15, mhi = (lane >> 4) * 8;
  float* o = out + (size_t)mat * BB * HID_;
#pragma unroll
  for (int r = 0; r < 8; ++r) {
    int col = c0 + nrel;
    o[(size_t)(r0 + r + mhi) * HID_ + col] = acc[r] + bias[col];
  }
}

extern "C" void kernel_launch(void* const* d_in, const int* in_sizes, int n_in,
                              void* d_out, int out_size, void* d_ws, size_t ws_size,
                              hipStream_t stream) {
  const float* x         = (const float*)d_in[0];
  const int*   ei        = (const int*)d_in[1];
  const float* edge_attr = (const float*)d_in[2];
  const int*   batch     = (const int*)d_in[3];
  const float* Wn        = (const float*)d_in[4];
  const float* bn        = (const float*)d_in[5];
  const float* We_enc    = (const float*)d_in[6];
  const float* be_enc    = (const float*)d_in[7];
  const float* gat_W     = (const float*)d_in[8];
  const float* gat_We    = (const float*)d_in[9];
  const float* att_src   = (const float*)d_in[10];
  const float* att_dst   = (const float*)d_in[11];
  const float* att_edge  = (const float*)d_in[12];
  const float* gat_b     = (const float*)d_in[13];
  const float* Wmu       = (const float*)d_in[14];
  const float* bmu       = (const float*)d_in[15];
  const float* Wlv       = (const float*)d_in[16];
  const float* blv       = (const float*)d_in[17];
  float* out = (float*)d_out;
  (void)in_sizes; (void)n_in; (void)out_size; (void)ws_size;

  char* ws = (char*)d_ws;
  size_t off = 0;
  auto alloc = [&](size_t elems) {
    float* p = (float*)(ws + off);
    off += elems * sizeof(float);
    return p;
  };
  float* h        = alloc((size_t)NN * HID_);
  float* hacc     = alloc((size_t)NN * HID_);
  float* xh       = alloc((size_t)NN * HID_);
  float* loop_sum = alloc((size_t)NN * HID_);
  float* deg      = alloc(NN);
  float* a_e      = alloc((size_t)LL * EA_ * HEADS_);
  float* a_s      = alloc((size_t)NN * HEADS_);
  float* a_d      = alloc((size_t)NN * HEADS_);
  float* logit    = alloc((size_t)EA_ * HEADS_);
  float* segmax   = alloc((size_t)NN * HEADS_);
  float* segsum   = alloc((size_t)NN * HEADS_);
  float* weff     = alloc((size_t)LL * 3 * HEADS_ * HID_);
  float* pooled   = alloc((size_t)BB * HID_);
  float* cnt      = alloc((size_t)BB);
  float* pnorm    = alloc((size_t)BB * HID_);
  unsigned short* hbf  = (unsigned short*)alloc((size_t)NN * HID_ / 2);
  unsigned short* wbf  = (unsigned short*)alloc((size_t)LL * HID_ * HID_ / 2);
  unsigned short* xbf  = (unsigned short*)alloc((size_t)NN * FN / 2);
  unsigned short* eabf = (unsigned short*)alloc((size_t)EE * FE / 2);
  unsigned short* wnbf = (unsigned short*)alloc((size_t)HID_ * FN / 2);
  unsigned short* webf = (unsigned short*)alloc((size_t)HID_ * FE / 2);

  auto nb = [](size_t n, int b) { return (unsigned)((n + (size_t)b - 1) / (size_t)b); };

  k_fill<<<nb((size_t)NN * HID_, 256), 256, 0, stream>>>(loop_sum, 0.f, (size_t)NN * HID_);
  k_fill<<<nb(NN, 256), 256, 0, stream>>>(deg, 0.f, NN);
  k_fill<<<nb((size_t)BB * HID_, 256), 256, 0, stream>>>(pooled, 0.f, (size_t)BB * HID_);
  k_fill<<<nb(BB, 256), 256, 0, stream>>>(cnt, 0.f, BB);

  k_w2bf<<<nb(LL * HID_ * HID_, 256), 256, 0, stream>>>(gat_W, wbf, LL * HID_ * HID_);
  k_w2bf<<<nb((size_t)NN * FN, 256), 256, 0, stream>>>(x, xbf, NN * FN);
  k_w2bf<<<nb((size_t)EE * FE, 256), 256, 0, stream>>>(edge_attr, eabf, EE * FE);
  k_w2bf<<<nb(HID_ * FN, 256), 256, 0, stream>>>(Wn, wnbf, HID_ * FN);
  k_w2bf<<<nb(HID_ * FE, 256), 256, 0, stream>>>(We_enc, webf, HID_ * FE);
  k_weff<<<nb(LL * 3 * HEADS_ * HID_, 256), 256, 0, stream>>>(gat_W, gat_We, att_src,
                                                              att_dst, att_edge, weff);
  k_node_enc<<<NN / 16, 256, 0, stream>>>(xbf, wnbf, bn, h, hbf);
  k_edge_enc<<<EE / 16, 256, 0, stream>>>(eabf, ei, webf, be_enc, weff, a_e,
                                          loop_sum, deg);
  k_selfloop<<<nb((size_t)NN * 12, 256), 256, 0, stream>>>(loop_sum, deg, weff, a_e);

  for (int l = 0; l < LL; ++l) {
    k_fill<<<nb((size_t)NN * HEADS_, 256), 256, 0, stream>>>(segmax, -1e30f,
                                                             (size_t)NN * HEADS_);
    k_fill<<<nb((size_t)NN * HEADS_, 256), 256, 0, stream>>>(segsum, 0.f,
                                                             (size_t)NN * HEADS_);
    k_fill<<<nb((size_t)NN * HID_, 256), 256, 0, stream>>>(hacc, 0.f, (size_t)NN * HID_);
    k_xh_attn<<<NN / 16, 256, 0, stream>>>(hbf, wbf + (size_t)l * HID_ * HID_, weff, l,
                                           xh, a_s, a_d);
    k_logit_max<<<nb((size_t)EA_ * HEADS_, 256), 256, 0, stream>>>(ei, a_s, a_d, a_e, l,
                                                                   logit, segmax);
    k_exp_sum<<<nb((size_t)EA_ * HEADS_, 256), 256, 0, stream>>>(ei, logit, segmax,
                                                                 segsum, xh);
    k_scatter<<<EA_, HID_, 0, stream>>>(ei, logit, segsum, xh, hacc);
    k_finish<<<nb((size_t)NN * HID_, 256), 256, 0, stream>>>(hacc, gat_b, l, h, hbf);
  }

  k_pool<<<NN, HID_, 0, stream>>>(h, batch, pooled, cnt);
  k_pnorm<<<nb((size_t)BB * HID_, 256), 256, 0, stream>>>(pooled, cnt, pnorm);
  k_head<<<8, 256, 0, stream>>>(pnorm, Wmu, bmu, Wlv, blv, out);
}
